// DABlock_704374637075
// MI455X (gfx1250) — matrix operand order repaired and checked by reading, not compiled, not run on hardware
//
#include <hip/hip_runtime.h>
#include <hip/hip_bf16.h>
#include <math.h>

typedef __attribute__((ext_vector_type(16))) _Float16 v16h;
typedef __attribute__((ext_vector_type(8)))  float    v8f;

#define NINST 100      // b*u*v = 4*5*5
#define CCH   64       // channels
#define HW    4096     // 64*64
#define CODE_C 16

__device__ __forceinline__ float leaky(float z) { return z >= 0.f ? z : 0.1f * z; }

// ---------------------------------------------------------------------------
// K1: generate per-instance depthwise 3x3 kernels; zero the sum accumulators.
// grid = 100 blocks, 64 threads.
// ---------------------------------------------------------------------------
__global__ __launch_bounds__(64)
void k1_gen_kernels(const float* __restrict__ code,   // (4,16,5,5)
                    const float* __restrict__ wg1,    // (64,16)
                    const float* __restrict__ wg2,    // (576,64)
                    float* __restrict__ ker,          // ws: (100,64,9)
                    float* __restrict__ sums)         // ws: (100,64) -> zeroed
{
    const int ni = blockIdx.x;
    const int t  = threadIdx.x;
    const int bi = ni / 25, rem = ni % 25, ui = rem / 5, vi = rem % 5;

    __shared__ float sc[CODE_C];
    __shared__ float sk1[64];

    if (t < CODE_C) sc[t] = code[((bi * CODE_C + t) * 5 + ui) * 5 + vi];
    sums[ni * CCH + t] = 0.f;
    __syncthreads();

    float a = 0.f;
    #pragma unroll
    for (int j = 0; j < CODE_C; ++j) a += wg1[t * CODE_C + j] * sc[j];
    sk1[t] = leaky(a);
    __syncthreads();

    #pragma unroll
    for (int q = 0; q < 9; ++q) {
        float acc = 0.f;
        const float* wr = wg2 + (t * 9 + q) * 64;
        #pragma unroll 8
        for (int j = 0; j < 64; ++j) acc += wr[j] * sk1[j];
        ker[(ni * CCH + t) * 9 + q] = acc;
    }
}

// ---------------------------------------------------------------------------
// K2: async-copy x rows to LDS -> depthwise 3x3 + leaky (f16 LDS tile) ->
//     WMMA 1x1 conv (64x64 GEMM per instance).
// grid = (32 pixel-tiles, 100 instances), 256 threads = 8 waves (wave32).
// Each block: 128 pixels (2 image rows) x 64 channels.
// Each wave: 16 pixels x all 64 output channels via 8x v_wmma_f32_16x16x32_f16.
// ---------------------------------------------------------------------------
#define FEA_STRIDE 72   // halves; 144B rows: 16B aligned, bank-conflict-friendly
#define W_STRIDE   72

__global__ __launch_bounds__(256)
void k2_dwconv_wmma(const float* __restrict__ x,       // (100,64,64,64)
                    const float* __restrict__ ker,     // ws (100,64,9)
                    const float* __restrict__ w1x1,    // (64,64) [o][i]
                    float* __restrict__ fea_da,        // ws (100,64,4096)
                    float* __restrict__ sums)          // ws (100,64)
{
    const int ni   = blockIdx.y;
    const int tile = blockIdx.x;        // 0..31, 128 pixels each
    const int t    = threadIdx.x;
    const int p0   = tile * 128;
    const int y0   = tile * 2;

    __shared__ float    sX[CCH * 4 * 64];        // x rows y0-1..y0+2, 64 KB
    __shared__ _Float16 sFea[128 * FEA_STRIDE];  // fea[p][i], f16
    __shared__ _Float16 sW[64 * W_STRIDE];       // W[o][i],  f16
    __shared__ float    sKer[64 * 9];
    __shared__ float    sSum[64];

    if (t < 64) sSum[t] = 0.f;
    for (int idx = t; idx < 576; idx += 256) sKer[idx] = ker[ni * 576 + idx];
    for (int idx = t; idx < 4096; idx += 256)
        sW[(idx >> 6) * W_STRIDE + (idx & 63)] = (_Float16)w1x1[idx];

    // ---- async global->LDS copy of 4 rows (zero-fill out-of-range rows) ----
    const float*   xbase   = x + (size_t)ni * CCH * HW;
    const unsigned sx_base = (unsigned)(uintptr_t)sX;
    #pragma unroll
    for (int r = 0; r < 4; ++r) {
        const int yy = y0 - 1 + r;
        if (yy >= 0 && yy < 64) {
            #pragma unroll
            for (int c4 = 0; c4 < 4; ++c4) {
                const int idx = c4 * 256 + t;          // 1024 float4 chunks
                const int ch  = idx >> 4;
                const int w4  = (idx & 15) << 2;
                const float*   gp  = xbase + (size_t)ch * HW + yy * 64 + w4;
                const unsigned lds = sx_base + ((((ch << 2) + r) << 6) + w4) * 4u;
                asm volatile("global_load_async_to_lds_b128 %0, %1, off"
                             :: "v"(lds), "v"(gp) : "memory");
            }
        } else {
            const float4 z = {0.f, 0.f, 0.f, 0.f};
            #pragma unroll
            for (int c4 = 0; c4 < 4; ++c4) {
                const int idx = c4 * 256 + t;
                const int ch  = idx >> 4;
                const int w4  = (idx & 15) << 2;
                *(float4*)&sX[(((ch << 2) + r) << 6) + w4] = z;
            }
        }
    }
    asm volatile("s_wait_asynccnt 0x0" ::: "memory");
    __syncthreads();

    // ---- depthwise 3x3 + leaky from LDS, 4-pixel microtiles ----------------
    // sX row r holds image row y0-1+r; output row `row` needs r = row..row+2.
    #pragma unroll
    for (int mm = 0; mm < 8; ++mm) {
        const int m   = mm * 256 + t;    // 2048 microtiles of 4 pixels
        const int i   = m >> 5;          // channel
        const int pm  = (m & 31) << 2;   // pixel start within 128
        const int row = pm >> 6;         // 0 or 1
        const int xx  = pm & 63;         // 0..60 step 4
        const float* kp = sKer + i * 9;
        float a0 = 0.f, a1 = 0.f, a2 = 0.f, a3 = 0.f;
        #pragma unroll
        for (int rr = 0; rr < 3; ++rr) {
            const float* xr = &sX[(((i << 2) + row + rr) << 6)];
            const float xm1 = (xx > 0)  ? xr[xx - 1] : 0.f;
            const float b0 = xr[xx], b1 = xr[xx + 1], b2 = xr[xx + 2], b3 = xr[xx + 3];
            const float xp4 = (xx < 60) ? xr[xx + 4] : 0.f;
            const float k0 = kp[rr * 3 + 0], k1 = kp[rr * 3 + 1], k2 = kp[rr * 3 + 2];
            a0 += k0 * xm1 + k1 * b0 + k2 * b1;
            a1 += k0 * b0  + k1 * b1 + k2 * b2;
            a2 += k0 * b1  + k1 * b2 + k2 * b3;
            a3 += k0 * b2  + k1 * b3 + k2 * xp4;
        }
        sFea[(pm + 0) * FEA_STRIDE + i] = (_Float16)leaky(a0);
        sFea[(pm + 1) * FEA_STRIDE + i] = (_Float16)leaky(a1);
        sFea[(pm + 2) * FEA_STRIDE + i] = (_Float16)leaky(a2);
        sFea[(pm + 3) * FEA_STRIDE + i] = (_Float16)leaky(a3);
    }
    __syncthreads();

    // ---- per-wave GEMM: out[o, p] = sum_i W[o][i] * fea[i][p] ---------------
    const int lane = t & 31;
    const int Lrow = lane & 15;         // M (A) / N (B,C/D) index
    const int Lhi  = lane >> 4;         // lane-group select
    const int pb   = (t >> 5) * 16;     // 16-pixel column tile per wave

    // A fragments: ISA 16-bit A 16x32 layout.
    v16h A[2][4];
    #pragma unroll
    for (int kc = 0; kc < 2; ++kc) {
        #pragma unroll
        for (int ot = 0; ot < 4; ++ot) {
            const _Float16* ap = &sW[(ot * 16 + Lrow) * W_STRIDE + kc * 32 + Lhi * 8];
            #pragma unroll
            for (int e = 0; e < 8; ++e) {
                A[kc][ot][e]     = ap[e];       // K = base + 0..7
                A[kc][ot][8 + e] = ap[16 + e];  // K = base + 16..23
            }
        }
    }

    v8f acc[4];
    #pragma unroll
    for (int ot = 0; ot < 4; ++ot)
        #pragma unroll
        for (int r = 0; r < 8; ++r) acc[ot][r] = 0.f;

    #pragma unroll
    for (int kc = 0; kc < 2; ++kc) {
        // B fragment: lane L holds N = L&15; halves 0..7 = K 8*(L>>4)+0..7, 8..15 = K 16+8*(L>>4)+0..7
        v16h B;
        const _Float16* bp = &sFea[(pb + Lrow) * FEA_STRIDE + kc * 32 + Lhi * 8];
        #pragma unroll
        for (int e = 0; e < 8; ++e) { B[e] = bp[e]; B[8 + e] = bp[16 + e]; }
        #pragma unroll
        for (int ot = 0; ot < 4; ++ot)
            acc[ot] = __builtin_amdgcn_wmma_f32_16x16x32_f16(
                false, A[kc][ot], false, B, (short)0, acc[ot], false, false);
    }

    // ---- epilogue: store fea_da, accumulate per-channel sums ---------------
    const int p = p0 + pb + Lrow;
    #pragma unroll
    for (int ot = 0; ot < 4; ++ot) {
        #pragma unroll
        for (int r = 0; r < 8; ++r) {
            const int o = ot * 16 + r + Lhi * 8;   // C/D layout: VGPR r, lane-hi -> M+8
            const float vv = acc[ot][r];
            fea_da[((size_t)ni * CCH + o) * HW + p] = vv;
            float s = vv;                           // reduce over the 16 pixels
            s += __shfl_xor(s, 1);
            s += __shfl_xor(s, 2);
            s += __shfl_xor(s, 4);
            s += __shfl_xor(s, 8);
            if (Lrow == 0) atomicAdd(&sSum[o], s);
        }
    }
    __syncthreads();
    if (t < 64) atomicAdd(&sums[ni * CCH + t], sSum[t]);
}

// ---------------------------------------------------------------------------
// K3: SE attention MLP -> attScale = 1 + sigmoid(...). grid = 100, 64 thr.
// ---------------------------------------------------------------------------
__global__ __launch_bounds__(64)
void k3_attention(const float* __restrict__ sums,   // (100,64)
                  const float* __restrict__ code,   // (4,16,5,5)
                  const float* __restrict__ wca1, const float* __restrict__ bca1,
                  const float* __restrict__ wca2, const float* __restrict__ bca2,
                  float* __restrict__ attScale)     // ws (100,64)
{
    const int ni = blockIdx.x;
    const int t  = threadIdx.x;
    const int bi = ni / 25, rem = ni % 25, ui = rem / 5, vi = rem % 5;

    __shared__ float cat[80];
    __shared__ float hid[16];

    cat[t] = sums[ni * CCH + t] * (1.f / 4096.f);
    if (t < CODE_C) cat[64 + t] = code[((bi * CODE_C + t) * 5 + ui) * 5 + vi];
    __syncthreads();

    if (t < 16) {
        float h = bca1[t];
        #pragma unroll 8
        for (int k = 0; k < 80; ++k) h += wca1[t * 80 + k] * cat[k];
        hid[t] = leaky(h);
    }
    __syncthreads();

    float a = bca2[t];
    #pragma unroll
    for (int j = 0; j < 16; ++j) a += wca2[t * 16 + j] * hid[j];
    attScale[ni * CCH + t] = 1.f + 1.f / (1.f + __expf(-a));
}

// ---------------------------------------------------------------------------
// K4: out = fea_da * attScale + x   (float4 vectorized). 25600 blocks x 256.
// ---------------------------------------------------------------------------
__global__ __launch_bounds__(256)
void k4_epilogue(const float* __restrict__ fea_da,
                 const float* __restrict__ attScale,
                 const float* __restrict__ x,
                 float* __restrict__ out)
{
    const int idx  = blockIdx.x * 256 + threadIdx.x;   // float4 index
    const int fidx = idx * 4;
    const int n = fidx >> 18;           // / (64*4096)
    const int i = (fidx >> 12) & 63;    // channel
    const float s = attScale[n * CCH + i];
    const float4 fd = ((const float4*)fea_da)[idx];
    const float4 xv = ((const float4*)x)[idx];
    float4 o;
    o.x = fd.x * s + xv.x;
    o.y = fd.y * s + xv.y;
    o.z = fd.z * s + xv.z;
    o.w = fd.w * s + xv.w;
    ((float4*)out)[idx] = o;
}

// ---------------------------------------------------------------------------
extern "C" void kernel_launch(void* const* d_in, const int* in_sizes, int n_in,
                              void* d_out, int out_size, void* d_ws, size_t ws_size,
                              hipStream_t stream) {
    const float* x     = (const float*)d_in[0];
    const float* code  = (const float*)d_in[1];
    const float* wg1   = (const float*)d_in[2];
    const float* wg2   = (const float*)d_in[3];
    const float* w1x1  = (const float*)d_in[4];
    const float* wca1  = (const float*)d_in[5];
    const float* bca1  = (const float*)d_in[6];
    const float* wca2  = (const float*)d_in[7];
    const float* bca2  = (const float*)d_in[8];
    float*       out   = (float*)d_out;

    // workspace layout (floats)
    float* wsf     = (float*)d_ws;
    float* fea_da  = wsf;                                    // 100*64*4096
    float* ker     = fea_da + (size_t)NINST * CCH * HW;      // 100*64*9
    float* sums    = ker + NINST * CCH * 9;                  // 100*64
    float* attS    = sums + NINST * CCH;                     // 100*64

    k1_gen_kernels<<<NINST, 64, 0, stream>>>(code, wg1, wg2, ker, sums);
    k2_dwconv_wmma<<<dim3(32, NINST), 256, 0, stream>>>(x, ker, w1x1, fea_da, sums);
    k3_attention<<<NINST, 64, 0, stream>>>(sums, code, wca1, bca1, wca2, bca2, attS);
    k4_epilogue<<<(NINST * CCH * HW) / (256 * 4), 256, 0, stream>>>(fea_da, attS, x, out);
}
